// PairwiseMessages_11931419148549
// MI455X (gfx1250) — compile-verified, matrix operand order checked by reading, not run on hardware
//
#include <hip/hip_runtime.h>
#include <math.h>

// Problem constants (from reference)
static constexpr int BB     = 2;
static constexpr int NQ     = 768;
static constexpr int NK     = 768;
static constexpr int D_EQUI = 256;
static constexpr int D_INV  = 256;
static constexpr int D_MSG  = 32;
static constexpr int D_FF   = 128;
static constexpr int D_OUT  = 64;

typedef __attribute__((ext_vector_type(16))) _Float16 v16h;
typedef __attribute__((ext_vector_type(8)))  _Float16 v8h;
typedef __attribute__((ext_vector_type(8)))  float    v8f;

static __device__ inline v16h cat8(v8h lo, v8h hi) {
    v16h r;
#pragma unroll
    for (int i = 0; i < 8; ++i) { r[i] = lo[i]; r[i + 8] = hi[i]; }
    return r;
}

// Generic pointer -> 32-bit LDS offset (for async-to-LDS VDST operand).
static __device__ inline unsigned lds_off(const void* p) {
    return (unsigned)(unsigned long long)
        (__attribute__((address_space(3))) const void*)p;
}

static __device__ inline v8f wmma16(v16h a, v16h b, v8f c) {
    return __builtin_amdgcn_wmma_f32_16x16x32_f16(false, a, false, b, (short)0, c,
                                                  false, false);
}

// gfx1250 hardware tanh TRANS op (confirmed lowering to v_tanh_f32).
static __device__ inline float fast_tanh(float u) {
#if __has_builtin(__builtin_amdgcn_tanhf)
    return __builtin_amdgcn_tanhf(u);
#else
    float e = __expf(2.f * u);
    return 1.f - 2.f * __builtin_amdgcn_rcpf(e + 1.f);
#endif
}

// ---------------------------------------------------------------------------
// Prep: [nrows x din] @ [din x 32] + bias -> f16.  One 32-lane group per row.
// ---------------------------------------------------------------------------
__global__ void proj_kernel(const float* __restrict__ X, const float* __restrict__ W,
                            const float* __restrict__ bias, _Float16* __restrict__ Y,
                            int nrows, int din) {
    int r = blockIdx.x * 8 + (threadIdx.x >> 5);
    int m = threadIdx.x & 31;
    if (r >= nrows) return;
    const float* x = X + (size_t)r * din;
    float acc = bias[m];
    for (int j = 0; j < din; ++j) acc = fmaf(x[j], W[j * D_MSG + m], acc);
    Y[(size_t)r * D_MSG + m] = (_Float16)acc;
}

// Per-point squared norm over the 3 coordinate rows -> f16.
__global__ void norm_kernel(const _Float16* __restrict__ E, _Float16* __restrict__ N2h,
                            int npts) {
    int idx = blockIdx.x * 256 + threadIdx.x;
    if (idx >= npts * D_MSG) return;
    int row = idx / D_MSG, m = idx % D_MSG;
    float s = 0.f;
#pragma unroll
    for (int c = 0; c < 3; ++c) {
        float v = (float)E[((size_t)row * 3 + c) * D_MSG + m];
        s = fmaf(v, v, s);
    }
    N2h[idx] = (_Float16)s;
}

// Pre-transpose both MLP weights to f16 [out][in] (B-fragment friendly).
__global__ void wprep_kernel(const float* __restrict__ W1, const float* __restrict__ W2,
                             _Float16* __restrict__ w1t, _Float16* __restrict__ w2t) {
    int i = blockIdx.x * 256 + threadIdx.x;
    if (i < D_FF * D_FF) {
        int n = i >> 7, j = i & 127;
        w1t[i] = (_Float16)W1[j * D_FF + n];
    }
    if (i < D_OUT * D_FF) {
        int n = i >> 7, j = i & 127;
        w2t[i] = (_Float16)W2[j * D_OUT + n];
    }
}

// Per-k feature fragments (ki broadcast, packed-f16 dot, norm-trick dist).
struct KFeat { v16h ak, adot, adist; };
static __device__ inline KFeat make_kfeat(const _Float16* __restrict__ kiH,
                                          const _Float16* __restrict__ keH,
                                          const _Float16* __restrict__ nk2g,
                                          size_t koff, int mb,
                                          const v16h* qe, v16h nq2h) {
    KFeat f;
    f.ak = cat8(*(const v8h*)(kiH + koff * D_MSG + mb),
                *(const v8h*)(kiH + koff * D_MSG + mb + 16));
    v16h ke0 = cat8(*(const v8h*)(keH + (koff * 3 + 0) * D_MSG + mb),
                    *(const v8h*)(keH + (koff * 3 + 0) * D_MSG + mb + 16));
    f.adot = qe[0] * ke0;
#pragma unroll
    for (int c = 1; c < 3; ++c) {
        v16h kec = cat8(*(const v8h*)(keH + (koff * 3 + c) * D_MSG + mb),
                        *(const v8h*)(keH + (koff * 3 + c) * D_MSG + mb + 16));
        f.adot += qe[c] * kec;
    }
    v16h nk2h = cat8(*(const v8h*)(nk2g + koff * D_MSG + mb),
                     *(const v8h*)(nk2g + koff * D_MSG + mb + 16));
    v16h s = nq2h + nk2h - (f.adot + f.adot);
#pragma unroll
    for (int i = 0; i < 16; ++i) {
        _Float16 si = s[i] > (_Float16)0.f ? s[i] : (_Float16)0.f;
        f.adist[i] = __builtin_sqrtf16(si);
    }
    return f;
}

// ---------------------------------------------------------------------------
// Main pairwise kernel. Grid (96 q-tiles, 8 k-chunks), 256 threads (8 waves).
// Each wave handles TWO 16-pair row blocks per iteration so every LDS weight
// fragment feeds two WMMAs.
// ---------------------------------------------------------------------------
__launch_bounds__(256)
__global__ void pair_kernel(const _Float16* __restrict__ qiH, const _Float16* __restrict__ kiH,
                            const _Float16* __restrict__ qeH, const _Float16* __restrict__ keH,
                            const _Float16* __restrict__ nq2g, const _Float16* __restrict__ nk2g,
                            const _Float16* __restrict__ w1tG, const _Float16* __restrict__ w2tG,
                            const float* __restrict__ b1g, const float* __restrict__ b2g,
                            float* __restrict__ out) {
    __shared__ _Float16 w1t[D_FF * D_FF];        // [n][k] = W1[k][n]
    __shared__ _Float16 w2t[D_OUT * D_FF];       // [n][k] = W2[k][n]
    __shared__ _Float16 hbuf[8][2][16 * D_FF];   // per-wave, per-k h staging

    const int tid = threadIdx.x;

    // Stage f16 weights into LDS with CDNA5 async copies (ASYNCcnt path).
    {
        const char* g1 = (const char*)w1tG;
        const char* g2 = (const char*)w2tG;
        unsigned l1 = lds_off(&w1t[0]);
        unsigned l2 = lds_off(&w2t[0]);
#pragma unroll
        for (int i = 0; i < (D_FF * D_FF * 2) / (16 * 256); ++i) {   // 8 iters
            int e = (i * 256 + tid) * 16;
            asm volatile("global_load_async_to_lds_b128 %0, %1, off"
                         :: "v"(l1 + e), "v"(g1 + e) : "memory");
        }
#pragma unroll
        for (int i = 0; i < (D_OUT * D_FF * 2) / (16 * 256); ++i) {  // 4 iters
            int e = (i * 256 + tid) * 16;
            asm volatile("global_load_async_to_lds_b128 %0, %1, off"
                         :: "v"(l2 + e), "v"(g2 + e) : "memory");
        }
        asm volatile("s_wait_asynccnt 0x0" ::: "memory");
    }
    __syncthreads();

    const int w    = tid >> 5;        // wave in workgroup
    const int lane = tid & 31;
    const int g    = lane >> 4;       // lane group (ISA 16-bit A/B layout split)
    const int r    = lane & 15;       // row (A) / column (B,C,D) index
    const int mb   = g * 8;           // K-half base

    const int tq = blockIdx.x;                 // 0..95
    const int b  = tq / (NQ / 16);
    const int qb = (tq % (NQ / 16)) * 16;
    const size_t qoff = (size_t)b * NQ + (qb + r);

    // Feature block 0 (qi) A-fragment: constant over the k loop.
    const v16h aq = cat8(*(const v8h*)(qiH + qoff * D_MSG + mb),
                         *(const v8h*)(qiH + qoff * D_MSG + mb + 16));

    // Persistent packed-f16 equivariant values + norms (fragment channel order).
    v16h qe[3];
#pragma unroll
    for (int c = 0; c < 3; ++c)
        qe[c] = cat8(*(const v8h*)(qeH + (qoff * 3 + c) * D_MSG + mb),
                     *(const v8h*)(qeH + (qoff * 3 + c) * D_MSG + mb + 16));
    const v16h nq2h = cat8(*(const v8h*)(nq2g + qoff * D_MSG + mb),
                           *(const v8h*)(nq2g + qoff * D_MSG + mb + 16));

    // Biases (C/D layout fixes the column per lane): folded in post-GEMM adds.
    float b1v[8], b2v[4];
#pragma unroll
    for (int n = 0; n < 8; ++n) b1v[n] = b1g[n * 16 + r];
#pragma unroll
    for (int n = 0; n < 4; ++n) b2v[n] = b2g[n * 16 + r];

    _Float16* hrowA = &hbuf[w][0][0];
    _Float16* hrowB = &hbuf[w][1][0];
    const int kbase = blockIdx.y * (NK / 8) + w;

    for (int it = 0; it < 6; ++it) {
        const int kA = kbase + it * 16;
        const int kB = kA + 8;

        KFeat fA = make_kfeat(kiH, keH, nk2g, (size_t)b * NK + kA, mb, qe, nq2h);
        KFeat fB = make_kfeat(kiH, keH, nk2g, (size_t)b * NK + kB, mb, qe, nq2h);
        v16h afrA[4] = {aq, fA.ak, fA.adot, fA.adist};
        v16h afrB[4] = {aq, fB.ak, fB.adot, fB.adist};

        // ---- Layer 1 (n outermost; each B-fragment feeds both k blocks).
        v16h bfr[4];
#pragma unroll
        for (int kc = 0; kc < 4; ++kc) {
            const _Float16* wp = &w1t[r * D_FF + kc * 32 + mb];
            bfr[kc] = cat8(*(const v8h*)wp, *(const v8h*)(wp + 16));
        }
#pragma unroll
        for (int n = 0; n < 8; ++n) {
            v16h bnext[4];
            if (n < 7) {
#pragma unroll
                for (int kc = 0; kc < 4; ++kc) {
                    const _Float16* wp = &w1t[((n + 1) * 16 + r) * D_FF + kc * 32 + mb];
                    bnext[kc] = cat8(*(const v8h*)wp, *(const v8h*)(wp + 16));
                }
            }
            v8f accA = {};   // inline-0 C operand
            v8f accB = {};
#pragma unroll
            for (int kc = 0; kc < 4; ++kc) accA = wmma16(afrA[kc], bfr[kc], accA);
#pragma unroll
            for (int kc = 0; kc < 4; ++kc) accB = wmma16(afrB[kc], bfr[kc], accB);

            // SiLU via hardware tanh: silu(x) = u + u*tanh(u), u = x/2.
            float u[16], t[16];
#pragma unroll
            for (int v = 0; v < 8; ++v) {
                u[v]     = 0.5f * (accA[v] + b1v[n]);
                u[v + 8] = 0.5f * (accB[v] + b1v[n]);
            }
#pragma unroll
            for (int i = 0; i < 16; ++i) t[i] = fast_tanh(u[i]);
#pragma unroll
            for (int v = 0; v < 8; ++v) {
                int row = v + 8 * g, col = n * 16 + r;
                hrowA[row * D_FF + col] = (_Float16)fmaf(u[v], t[v], u[v]);
                hrowB[row * D_FF + col] = (_Float16)fmaf(u[v + 8], t[v + 8], u[v + 8]);
            }
#pragma unroll
            for (int kc = 0; kc < 4; ++kc) bfr[kc] = bnext[kc];
        }
        // Wave-local DS ordering: h stores must land before cross-lane reads.
        asm volatile("s_wait_dscnt 0x0" ::: "memory");

        v16h hfA[4], hfB[4];
#pragma unroll
        for (int kc = 0; kc < 4; ++kc) {
            const _Float16* hpA = &hrowA[r * D_FF + kc * 32 + mb];
            const _Float16* hpB = &hrowB[r * D_FF + kc * 32 + mb];
            hfA[kc] = cat8(*(const v8h*)hpA, *(const v8h*)(hpA + 16));
            hfB[kc] = cat8(*(const v8h*)hpB, *(const v8h*)(hpB + 16));
        }

        // ---- Layer 2 (n2 outermost; W2 fragments shared by both k blocks).
        float* opA = out + (((size_t)b * NQ + qb) * NK + kA) * D_OUT;
        float* opB = opA + (size_t)8 * D_OUT;
        v16h b2f[4];
#pragma unroll
        for (int kc = 0; kc < 4; ++kc) {
            const _Float16* wp = &w2t[r * D_FF + kc * 32 + mb];
            b2f[kc] = cat8(*(const v8h*)wp, *(const v8h*)(wp + 16));
        }
#pragma unroll
        for (int n2 = 0; n2 < 4; ++n2) {
            v16h b2n[4];
            if (n2 < 3) {
#pragma unroll
                for (int kc = 0; kc < 4; ++kc) {
                    const _Float16* wp = &w2t[((n2 + 1) * 16 + r) * D_FF + kc * 32 + mb];
                    b2n[kc] = cat8(*(const v8h*)wp, *(const v8h*)(wp + 16));
                }
            }
            v8f acc2A = {};
            v8f acc2B = {};
#pragma unroll
            for (int kc = 0; kc < 4; ++kc) acc2A = wmma16(hfA[kc], b2f[kc], acc2A);
#pragma unroll
            for (int kc = 0; kc < 4; ++kc) acc2B = wmma16(hfB[kc], b2f[kc], acc2B);

            // Non-temporal stores: 302 MB write-once stream must not churn L2.
#pragma unroll
            for (int v = 0; v < 8; ++v) {
                int row = v + 8 * g;
                __builtin_nontemporal_store(acc2A[v] + b2v[n2],
                                            &opA[(size_t)row * NK * D_OUT + n2 * 16 + r]);
                __builtin_nontemporal_store(acc2B[v] + b2v[n2],
                                            &opB[(size_t)row * NK * D_OUT + n2 * 16 + r]);
            }
#pragma unroll
            for (int kc = 0; kc < 4; ++kc) b2f[kc] = b2n[kc];
        }
    }
}

// ---------------------------------------------------------------------------
extern "C" void kernel_launch(void* const* d_in, const int* in_sizes, int n_in,
                              void* d_out, int out_size, void* d_ws, size_t ws_size,
                              hipStream_t stream) {
    const float* q_equi = (const float*)d_in[0];
    const float* q_inv  = (const float*)d_in[1];
    const float* k_equi = (const float*)d_in[2];
    const float* k_inv  = (const float*)d_in[3];
    const float* Wqi = (const float*)d_in[4];   const float* bqi = (const float*)d_in[5];
    const float* Wki = (const float*)d_in[6];   const float* bki = (const float*)d_in[7];
    const float* Wqe = (const float*)d_in[8];   const float* bqe = (const float*)d_in[9];
    const float* Wke = (const float*)d_in[10];  const float* bke = (const float*)d_in[11];
    const float* W1  = (const float*)d_in[12];  const float* b1  = (const float*)d_in[13];
    const float* W2  = (const float*)d_in[14];  const float* b2  = (const float*)d_in[15];
    float* out = (float*)d_out;

    // Workspace layout (bytes, 16B-aligned regions)
    char* ws = (char*)d_ws;
    _Float16* qiH  = (_Float16*)(ws + 0);        //  B*NQ*32   f16 =  98304 B
    _Float16* kiH  = (_Float16*)(ws + 98304);    //  B*NK*32   f16
    _Float16* qeH  = (_Float16*)(ws + 196608);   //  B*NQ*3*32 f16 = 294912 B
    _Float16* keH  = (_Float16*)(ws + 491520);   //  B*NK*3*32 f16
    _Float16* nq2h = (_Float16*)(ws + 786432);   //  B*NQ*32   f16 =  98304 B
    _Float16* nk2h = (_Float16*)(ws + 884736);   //  B*NK*32   f16
    _Float16* w1tG = (_Float16*)(ws + 983040);   //  128*128   f16 =  32768 B
    _Float16* w2tG = (_Float16*)(ws + 1015808);  //  64*128    f16 =  16384 B

    const int rows_i = BB * NQ;        // 1536
    const int rows_e = BB * NQ * 3;    // 4608

    proj_kernel<<<dim3(rows_i / 8), dim3(256), 0, stream>>>(q_inv, Wqi, bqi, qiH, rows_i, D_INV);
    proj_kernel<<<dim3(rows_i / 8), dim3(256), 0, stream>>>(k_inv, Wki, bki, kiH, rows_i, D_INV);
    proj_kernel<<<dim3(rows_e / 8), dim3(256), 0, stream>>>(q_equi, Wqe, bqe, qeH, rows_e, D_EQUI);
    proj_kernel<<<dim3(rows_e / 8), dim3(256), 0, stream>>>(k_equi, Wke, bke, keH, rows_e, D_EQUI);
    norm_kernel<<<dim3(rows_i * D_MSG / 256), dim3(256), 0, stream>>>(qeH, nq2h, rows_i);
    norm_kernel<<<dim3(rows_i * D_MSG / 256), dim3(256), 0, stream>>>(keH, nk2h, rows_i);
    wprep_kernel<<<dim3(D_FF * D_FF / 256), dim3(256), 0, stream>>>(W1, W2, w1tG, w2tG);

    pair_kernel<<<dim3(BB * NQ / 16, 8), dim3(256), 0, stream>>>(
        qiH, kiH, qeH, keH, nq2h, nk2h, w1tG, w2tG, b1, b2, out);
}
